// TSA_Block_37701222924652
// MI455X (gfx1250) — matrix operand order repaired and checked by reading, not compiled, not run on hardware
//
#include <hip/hip_runtime.h>

// ---------------------------------------------------------------------------
// Self-attention block for MI455X (gfx1250, wave32, WMMA + async LDS DMA).
// GEMMs: v_wmma_f32_16x16x32_f16, tiles staged with global_load_async_to_lds.
// ---------------------------------------------------------------------------

typedef _Float16 h8   __attribute__((ext_vector_type(8)));
typedef _Float16 v16h __attribute__((ext_vector_type(16)));
typedef float    v8f  __attribute__((ext_vector_type(8)));

#define B_DIM 4
#define S_DIM 2048
#define D_DIM 1024

union HF16x16 { v16h v; h8 h[2]; };

// generic -> LDS(addrspace 3) offset for async-to-LDS VDST operand
typedef __attribute__((address_space(3))) const _Float16 lds_cf16;
__device__ __forceinline__ unsigned lds_off(const _Float16* p) {
    return (unsigned)(size_t)(lds_cf16*)p;
}

// CDNA5 async DMA: memory -> LDS, 16 bytes per lane, tracked by ASYNCcnt.
__device__ __forceinline__ void async_b128(unsigned lds, const _Float16* g) {
    asm volatile("global_load_async_to_lds_b128 %0, %1, off"
                 :
                 : "v"(lds), "v"(g)
                 : "memory");
}
#define WAIT_ASYNC(imm) asm volatile("s_wait_asynccnt " imm ::: "memory")

// A fragment (16x32 f16, M x K). Per ISA 7.12.2:
// lanes 0-15: M=lane, halves 0..7 = K0..7, halves 8..15 = K16..23
// lanes 16-31: M=lane-16, halves 0..7 = K8..15, halves 8..15 = K24..31
__device__ __forceinline__ v16h a_frag(const _Float16* tile, int lane, int ld) {
    const _Float16* p = tile + (size_t)(lane & 15) * ld + ((lane >> 4) << 3);
    HF16x16 f;
    f.h[0] = *(const h8*)(p);
    f.h[1] = *(const h8*)(p + 16);
    return f.v;
}

// B fragment (32x16 f16, K x N): N = lane&15, halves 0..7 = K(8h+0..7), halves 8..15 = K(16+8h+0..7), h = lane>>4.
__device__ __forceinline__ v16h b_frag(const _Float16* tile, int lane, int ld) {
    const _Float16* p = tile + (size_t)((lane >> 4) << 3) * ld + (lane & 15);
    v16h f;
#pragma unroll
    for (int s = 0; s < 8; ++s) {
        f[s] = p[(size_t)s * ld];
        f[8 + s] = p[(size_t)(16 + s) * ld];
    }
    return f;
}

#define WMMA16(a, b, c) \
    __builtin_amdgcn_wmma_f32_16x16x32_f16(false, (a), false, (b), (short)0, (c), false, false)

// ---------------------------------------------------------------------------
// Kernel 1: h = x + positional encoding, cast to f16.
// ---------------------------------------------------------------------------
__global__ __launch_bounds__(256) void prep_h(const float* __restrict__ x,
                                              _Float16* __restrict__ H) {
    const int idx = blockIdx.x * 256 + threadIdx.x;
    const int d = idx & (D_DIM - 1);
    const int s = (idx >> 10) & (S_DIM - 1);
    float pe;
    if (d < D_DIM / 2) {
        float ang = (float)s * __powf(10000.f, -(2.f * (float)d) / (float)D_DIM);
        pe = sinf(ang);
    } else {
        float i2 = 2.f * (float)(d - D_DIM / 2);
        float ang = (float)s * __powf(10000.f, -i2 / (float)D_DIM);
        pe = cosf(ang);
    }
    H[idx] = (_Float16)(x[idx] + pe);
}

// ---------------------------------------------------------------------------
// Kernel 2: cast the three weight matrices to f16.
// ---------------------------------------------------------------------------
__global__ __launch_bounds__(256) void cast_w(const float* __restrict__ a,
                                              const float* __restrict__ b,
                                              const float* __restrict__ c,
                                              _Float16* __restrict__ ah,
                                              _Float16* __restrict__ bh,
                                              _Float16* __restrict__ ch) {
    const int i = blockIdx.x * 256 + threadIdx.x;
    ah[i] = (_Float16)a[i];
    bh[i] = (_Float16)b[i];
    ch[i] = (_Float16)c[i];
}

// ---------------------------------------------------------------------------
// Kernel 3: fused Q/K/V projection. 64x64 tile, double-buffered async LDS.
// All fragments preloaded, then 6 back-to-back WMMAs per K step per wave.
// ---------------------------------------------------------------------------
__global__ __launch_bounds__(256) void qkv_gemm(
    const _Float16* __restrict__ H,
    const _Float16* __restrict__ Wq, const _Float16* __restrict__ Wk,
    const _Float16* __restrict__ Wv,
    const float* __restrict__ bq, const float* __restrict__ bk,
    const float* __restrict__ bv,
    _Float16* __restrict__ Q, _Float16* __restrict__ K,
    _Float16* __restrict__ V) {
    __shared__ __align__(16) _Float16 As[2][64 * 32];
    __shared__ __align__(16) _Float16 Bs[2][3][32 * 64];

    const int t = threadIdx.x;
    const int lane = t & 31, w = t >> 5;
    const int wm = w & 3, wn = w >> 2;
    const int m0 = blockIdx.x * 64;     // rows of (B*S)
    const int n0 = blockIdx.y * 64;     // output features

    v8f acc[3][2] = {};

    const int arow = t >> 2, acb = (t & 3) << 3;  // 64 rows x 4 chunks of 8
    const int brow = t >> 3, bcb = (t & 7) << 3;  // 32 rows x 8 chunks of 8

    const _Float16* gA  = H  + (size_t)(m0 + arow) * D_DIM + acb;
    const _Float16* gB0 = Wq + (size_t)brow * D_DIM + n0 + bcb;
    const _Float16* gB1 = Wk + (size_t)brow * D_DIM + n0 + bcb;
    const _Float16* gB2 = Wv + (size_t)brow * D_DIM + n0 + bcb;

    unsigned lA[2], lB[2][3];
#pragma unroll
    for (int p = 0; p < 2; ++p) {
        lA[p]    = lds_off(&As[p][arow * 32 + acb]);
        lB[p][0] = lds_off(&Bs[p][0][brow * 64 + bcb]);
        lB[p][1] = lds_off(&Bs[p][1][brow * 64 + bcb]);
        lB[p][2] = lds_off(&Bs[p][2][brow * 64 + bcb]);
    }

    // prologue: tile 0 -> buffer 0 (4 async instructions per wave)
    async_b128(lA[0], gA);
    async_b128(lB[0][0], gB0);
    async_b128(lB[0][1], gB1);
    async_b128(lB[0][2], gB2);

    const int KT = D_DIM / 32;
    for (int i = 0; i < KT; ++i) {
        const int cur = i & 1;
        if (i + 1 < KT) {
            const size_t k = (size_t)(i + 1) * 32;
            async_b128(lA[1 - cur], gA + k);
            async_b128(lB[1 - cur][0], gB0 + k * D_DIM);
            async_b128(lB[1 - cur][1], gB1 + k * D_DIM);
            async_b128(lB[1 - cur][2], gB2 + k * D_DIM);
            WAIT_ASYNC("0x4");   // tile i resident (in-order completion)
        } else {
            WAIT_ASYNC("0x0");
        }
        __syncthreads();

        // preload ALL fragments, then issue the WMMA chain back-to-back
        v16h a = a_frag(&As[cur][wm * 16 * 32], lane, 32);
        v16h bf[3][2];
#pragma unroll
        for (int h = 0; h < 2; ++h) {
            const int nb = wn * 32 + h * 16;
            bf[0][h] = b_frag(&Bs[cur][0][nb], lane, 64);
            bf[1][h] = b_frag(&Bs[cur][1][nb], lane, 64);
            bf[2][h] = b_frag(&Bs[cur][2][nb], lane, 64);
        }
#pragma unroll
        for (int h = 0; h < 2; ++h) {
            acc[0][h] = WMMA16(a, bf[0][h], acc[0][h]);
            acc[1][h] = WMMA16(a, bf[1][h], acc[1][h]);
            acc[2][h] = WMMA16(a, bf[2][h], acc[2][h]);
        }
        __syncthreads();
    }

#pragma unroll
    for (int h = 0; h < 2; ++h) {
        const int n = n0 + wn * 32 + h * 16 + (lane & 15);
        const float bQ = bq[n], bK = bk[n], bV = bv[n];
#pragma unroll
        for (int r = 0; r < 8; ++r) {
            const int m = m0 + wm * 16 + r + ((lane >> 4) << 3);
            const size_t o = (size_t)m * D_DIM + n;
            Q[o] = (_Float16)(acc[0][h][r] + bQ);
            K[o] = (_Float16)(acc[1][h][r] + bK);
            V[o] = (_Float16)(acc[2][h][r] + bV);
        }
    }
}

// ---------------------------------------------------------------------------
// Kernel 4: logits = Q K^T / sqrt(D) per batch. 64x128 tile; Q tile via async
// DMA, K tile loaded in registers and transposed through LDS (pipelined).
// Wave owns 16x64 -> 4 WMMAs per A fragment. Output f16.
// ---------------------------------------------------------------------------
__global__ __launch_bounds__(256) void logits_gemm(
    const _Float16* __restrict__ Q, const _Float16* __restrict__ Kg,
    _Float16* __restrict__ L) {
    __shared__ __align__(16) _Float16 As[2][64 * 32];
    __shared__ __align__(16) _Float16 Bs[2][32 * 128];

    const int b = blockIdx.z;
    const _Float16* Qb = Q + (size_t)b * S_DIM * D_DIM;
    const _Float16* Kb = Kg + (size_t)b * S_DIM * D_DIM;
    _Float16* Lb = L + (size_t)b * S_DIM * S_DIM;

    const int t = threadIdx.x;
    const int lane = t & 31, w = t >> 5;
    const int wm = w & 3, wn = w >> 2;         // wn in 0..1 -> 64 keys each
    const int m0 = blockIdx.x * 64;            // query rows
    const int n0 = blockIdx.y * 128;           // key columns

    v8f acc[4] = {};

    const int arow = t >> 2, acb = (t & 3) << 3;
    const int key = t >> 1, f0 = (t & 1) << 4; // 128 keys x 2 chunks of 16 feat

    const _Float16* gA = Qb + (size_t)(m0 + arow) * D_DIM + acb;
    const _Float16* gK = Kb + (size_t)(n0 + key) * D_DIM + f0;

    unsigned lA[2] = { lds_off(&As[0][arow * 32 + acb]),
                       lds_off(&As[1][arow * 32 + acb]) };

    // prologue
    async_b128(lA[0], gA);
    h8 kv0 = *(const h8*)(gK);
    h8 kv1 = *(const h8*)(gK + 8);

    const int KT = D_DIM / 32;
    for (int i = 0; i < KT; ++i) {
        const int cur = i & 1;
        // store current K tile transposed: Bs[feat][key]
#pragma unroll
        for (int j = 0; j < 8; ++j) Bs[cur][(f0 + j) * 128 + key] = kv0[j];
#pragma unroll
        for (int j = 0; j < 8; ++j) Bs[cur][(f0 + 8 + j) * 128 + key] = kv1[j];

        if (i + 1 < KT) {
            const size_t k = (size_t)(i + 1) * 32;
            async_b128(lA[1 - cur], gA + k);
            kv0 = *(const h8*)(gK + k);
            kv1 = *(const h8*)(gK + k + 8);
            WAIT_ASYNC("0x1");
        } else {
            WAIT_ASYNC("0x0");
        }
        __syncthreads();

        v16h a = a_frag(&As[cur][wm * 16 * 32], lane, 32);
        v16h bf[4];
#pragma unroll
        for (int h = 0; h < 4; ++h)
            bf[h] = b_frag(&Bs[cur][wn * 64 + h * 16], lane, 128);
#pragma unroll
        for (int h = 0; h < 4; ++h)
            acc[h] = WMMA16(a, bf[h], acc[h]);
        __syncthreads();
    }

    const float scale = 0.03125f;  // 1/sqrt(1024)
#pragma unroll
    for (int h = 0; h < 4; ++h) {
        const int n = n0 + wn * 64 + h * 16 + (lane & 15);
#pragma unroll
        for (int r = 0; r < 8; ++r) {
            const int m = m0 + wm * 16 + r + ((lane >> 4) << 3);
            Lb[(size_t)m * S_DIM + n] = (_Float16)(acc[h][r] * scale);
        }
    }
}

// ---------------------------------------------------------------------------
// Kernel 5: row softmax in place (f16 logits -> f16 probs).
// ---------------------------------------------------------------------------
__global__ __launch_bounds__(256) void softmax_rows(_Float16* __restrict__ L) {
    __shared__ float red[256];
    const int t = threadIdx.x;
    _Float16* p = L + (size_t)blockIdx.x * S_DIM;

    float v[8];
    float m = -1e30f;
#pragma unroll
    for (int i = 0; i < 8; ++i) {
        v[i] = (float)p[t + i * 256];
        m = fmaxf(m, v[i]);
    }
    red[t] = m;
    __syncthreads();
    for (int s = 128; s > 0; s >>= 1) {
        if (t < s) red[t] = fmaxf(red[t], red[t + s]);
        __syncthreads();
    }
    m = red[0];
    __syncthreads();

    float sum = 0.f;
#pragma unroll
    for (int i = 0; i < 8; ++i) {
        v[i] = __expf(v[i] - m);
        sum += v[i];
    }
    red[t] = sum;
    __syncthreads();
    for (int s = 128; s > 0; s >>= 1) {
        if (t < s) red[t] += red[t + s];
        __syncthreads();
    }
    const float inv = 1.f / red[0];
#pragma unroll
    for (int i = 0; i < 8; ++i) p[t + i * 256] = (_Float16)(v[i] * inv);
}

// ---------------------------------------------------------------------------
// Kernel 6: out = P * V + x (fused residual), f32 output. 64x128 tile,
// fully async double-buffered. K dim = S.
// ---------------------------------------------------------------------------
__global__ __launch_bounds__(256) void pv_gemm(
    const _Float16* __restrict__ P, const _Float16* __restrict__ V,
    const float* __restrict__ x, float* __restrict__ out) {
    __shared__ __align__(16) _Float16 As[2][64 * 32];
    __shared__ __align__(16) _Float16 Bs[2][32 * 128];

    const int b = blockIdx.z;
    const _Float16* Pb = P + (size_t)b * S_DIM * S_DIM;
    const _Float16* Vb = V + (size_t)b * S_DIM * D_DIM;

    const int t = threadIdx.x;
    const int lane = t & 31, w = t >> 5;
    const int wm = w & 3, wn = w >> 2;        // wn in 0..1 -> 64 features each
    const int m0 = blockIdx.x * 64;           // query rows
    const int n0 = blockIdx.y * 128;          // output features

    v8f acc[4] = {};

    const int arow = t >> 2, acb = (t & 3) << 3;
    const int brow = t >> 3, bcb = (t & 7) << 4;  // 32 rows x 8 chunks of 16

    const _Float16* gA = Pb + (size_t)(m0 + arow) * S_DIM + acb;
    const _Float16* gB = Vb + (size_t)brow * D_DIM + n0 + bcb;

    unsigned lA[2], lB[2];
#pragma unroll
    for (int p = 0; p < 2; ++p) {
        lA[p] = lds_off(&As[p][arow * 32 + acb]);
        lB[p] = lds_off(&Bs[p][brow * 128 + bcb]);
    }

    // prologue: tile 0 (3 async instructions: 1 A + 2 B halves)
    async_b128(lA[0], gA);
    async_b128(lB[0], gB);
    async_b128(lB[0] + 16, gB + 8);

    const int KT = S_DIM / 32;
    for (int i = 0; i < KT; ++i) {
        const int cur = i & 1;
        if (i + 1 < KT) {
            const size_t k = (size_t)(i + 1) * 32;
            async_b128(lA[1 - cur], gA + k);
            async_b128(lB[1 - cur], gB + k * D_DIM);
            async_b128(lB[1 - cur] + 16, gB + k * D_DIM + 8);
            WAIT_ASYNC("0x3");
        } else {
            WAIT_ASYNC("0x0");
        }
        __syncthreads();

        v16h a = a_frag(&As[cur][wm * 16 * 32], lane, 32);
        v16h bf[4];
#pragma unroll
        for (int h = 0; h < 4; ++h)
            bf[h] = b_frag(&Bs[cur][wn * 64 + h * 16], lane, 128);
#pragma unroll
        for (int h = 0; h < 4; ++h)
            acc[h] = WMMA16(a, bf[h], acc[h]);
        __syncthreads();
    }

#pragma unroll
    for (int h = 0; h < 4; ++h) {
        const int n = n0 + wn * 64 + h * 16 + (lane & 15);
#pragma unroll
        for (int r = 0; r < 8; ++r) {
            const int m = m0 + wm * 16 + r + ((lane >> 4) << 3);
            const size_t o = ((size_t)b * S_DIM + m) * D_DIM + n;
            out[o] = acc[h][r] + x[o];
        }
    }
}

// ---------------------------------------------------------------------------
extern "C" void kernel_launch(void* const* d_in, const int* in_sizes, int n_in,
                              void* d_out, int out_size, void* d_ws, size_t ws_size,
                              hipStream_t stream) {
    const float* x  = (const float*)d_in[0];
    const float* Wq = (const float*)d_in[1];
    const float* bq = (const float*)d_in[2];
    const float* Wk = (const float*)d_in[3];
    const float* bk = (const float*)d_in[4];
    const float* Wv = (const float*)d_in[5];
    const float* bv = (const float*)d_in[6];
    float* out = (float*)d_out;

    char* ws = (char*)d_ws;
    size_t off = 0;
    const size_t BSD = (size_t)B_DIM * S_DIM * D_DIM;
    const size_t DD  = (size_t)D_DIM * D_DIM;
    const size_t BSS = (size_t)B_DIM * S_DIM * S_DIM;

    _Float16* H   = (_Float16*)(ws + off); off += BSD * 2;
    _Float16* Wqh = (_Float16*)(ws + off); off += DD * 2;
    _Float16* Wkh = (_Float16*)(ws + off); off += DD * 2;
    _Float16* Wvh = (_Float16*)(ws + off); off += DD * 2;
    _Float16* Qh  = (_Float16*)(ws + off); off += BSD * 2;
    _Float16* Kh  = (_Float16*)(ws + off); off += BSD * 2;
    _Float16* Vh  = (_Float16*)(ws + off); off += BSD * 2;
    _Float16* L   = (_Float16*)(ws + off); off += BSS * 2;  // logits -> probs

    prep_h<<<(unsigned)(BSD / 256), 256, 0, stream>>>(x, H);
    cast_w<<<(unsigned)(DD / 256), 256, 0, stream>>>(Wq, Wk, Wv, Wqh, Wkh, Wvh);
    {
        dim3 grid((B_DIM * S_DIM) / 64, D_DIM / 64, 1);
        qkv_gemm<<<grid, 256, 0, stream>>>(H, Wqh, Wkh, Wvh, bq, bk, bv,
                                           Qh, Kh, Vh);
    }
    {
        dim3 grid(S_DIM / 64, S_DIM / 128, B_DIM);
        logits_gemm<<<grid, 256, 0, stream>>>(Qh, Kh, L);
    }
    softmax_rows<<<B_DIM * S_DIM, 256, 0, stream>>>(L);
    {
        dim3 grid(S_DIM / 64, D_DIM / 128, B_DIM);
        pv_gemm<<<grid, 256, 0, stream>>>(L, Vh, x, out);
    }
}
